// MaskCrossAttention_54056458388099
// MI455X (gfx1250) — matrix operand order repaired and checked by reading, not compiled, not run on hardware
//
#include <hip/hip_runtime.h>
#include <hip/hip_bf16.h>

typedef __attribute__((ext_vector_type(16))) _Float16 v16h;
typedef __attribute__((ext_vector_type(8)))  _Float16 v8h;
typedef __attribute__((ext_vector_type(8)))  float    v8f;

// B=8, C=512, H=W=64, HEADS=8, HD=64, HR=WR=32, S=1024
static constexpr float SCALE = 0.125f; // 64^-0.5, folded into q projection

// ---- workspace layout (bytes) ----
static constexpr size_t SZ_ACT_H = 8ull * 1024 * 512 * 2;
static constexpr size_t OFF_XR   = 0;
static constexpr size_t OFF_ZR   = OFF_XR + SZ_ACT_H;
static constexpr size_t OFF_Q    = OFF_ZR + SZ_ACT_H;
static constexpr size_t OFF_K    = OFF_Q  + SZ_ACT_H;
static constexpr size_t OFF_VT   = OFF_K  + SZ_ACT_H;
static constexpr size_t OFF_AO   = OFF_VT + SZ_ACT_H;
static constexpr size_t OFF_PROJ = OFF_AO + SZ_ACT_H;             // f32 [b][c][p]
static constexpr size_t OFF_MR   = OFF_PROJ + 8ull*512*1024*4;    // f32 [b][p]
static constexpr size_t OFF_W    = OFF_MR + 8ull*1024*4;          // f16 4x[512][512]

__device__ __forceinline__ v16h cat16(v8h lo, v8h hi) {
    return __builtin_shufflevector(lo, hi, 0,1,2,3,4,5,6,7,8,9,10,11,12,13,14,15);
}
__device__ __forceinline__ v8f wmma_f16(v16h a, v16h b, v8f c) {
    return __builtin_amdgcn_wmma_f32_16x16x32_f16(false, a, false, b, (short)0, c, false, false);
}
__device__ __forceinline__ float redmax16(float v) {
    v = fmaxf(v, __shfl_xor(v, 1, 32));
    v = fmaxf(v, __shfl_xor(v, 2, 32));
    v = fmaxf(v, __shfl_xor(v, 4, 32));
    v = fmaxf(v, __shfl_xor(v, 8, 32));
    return v;
}
__device__ __forceinline__ float redsum16(float v) {
    v += __shfl_xor(v, 1, 32);
    v += __shfl_xor(v, 2, 32);
    v += __shfl_xor(v, 4, 32);
    v += __shfl_xor(v, 8, 32);
    return v;
}

// CDNA5 async copy: 16B global -> LDS, tracked by ASYNCcnt.
// Generic LDS pointers truncate to the DS byte offset (aperture rule), so the
// low 32 bits are a valid VDST LDS address for the async-to-LDS op.
__device__ __forceinline__ void async_cp16(void* lds, const void* g) {
    unsigned l = (unsigned)(unsigned long long)lds;
    asm volatile("global_load_async_to_lds_b128 %0, %1, off"
                 :: "v"(l), "v"(g) : "memory");
}
__device__ __forceinline__ void async_wait0() {
    asm volatile("s_wait_asynccnt 0x0" ::: "memory");
}

// ---- 2x2 avg pool, fp32 [b][c][64][64] -> f16 channel-last [b][p][c] ----
__global__ void pool_f16_kernel(const float* __restrict__ src, _Float16* __restrict__ dst) {
    int idx = blockIdx.x * blockDim.x + threadIdx.x;
    if (idx >= 8 * 512 * 32 * 32) return;
    int wr = idx & 31, hr = (idx >> 5) & 31, c = (idx >> 10) & 511, b = idx >> 19;
    const float* p = src + (((size_t)(b * 512 + c) * 64 + 2 * hr) * 64 + 2 * wr);
    float2 a0 = *(const float2*)p;
    float2 a1 = *(const float2*)(p + 64);
    float v = 0.25f * (a0.x + a0.y + a1.x + a1.y);
    dst[((size_t)b * 1024 + (hr * 32 + wr)) * 512 + c] = (_Float16)v;
}

// ---- mask 2x2 avg (== bilinear half-res, align_corners=False) ----
__global__ void pool_mask_kernel(const float* __restrict__ m, float* __restrict__ mr) {
    int idx = blockIdx.x * blockDim.x + threadIdx.x;
    if (idx >= 8 * 1024) return;
    int wr = idx & 31, hr = (idx >> 5) & 31, b = idx >> 10;
    const float* p = m + (((size_t)b * 64 + 2 * hr) * 64 + 2 * wr);
    float2 a0 = *(const float2*)p;
    float2 a1 = *(const float2*)(p + 64);
    mr[idx] = 0.25f * (a0.x + a0.y + a1.x + a1.y);
}

// ---- convert 4 weight matrices fp32 -> f16 ----
__global__ void cvt_w_kernel(const float* __restrict__ wq, const float* __restrict__ wk,
                             const float* __restrict__ wv, const float* __restrict__ wo,
                             _Float16* __restrict__ dst) {
    int idx = blockIdx.x * blockDim.x + threadIdx.x;
    if (idx >= 4 * 512 * 512) return;
    int which = idx >> 18, i = idx & 262143;
    const float* s = (which == 0) ? wq : (which == 1) ? wk : (which == 2) ? wv : wo;
    dst[idx] = (_Float16)s[i];
}

// ---- tiled GEMM: out[o][p] = sum_c w[o][c]*act[b][p][c] + bias[o]  (then *oscale) ----
// MODE 0: f16 pixel-major out[b][p][o]   (q,k; oscale folds attention scale into q)
// MODE 1: f16 channel-major out[b][o][p] (v transposed)
// MODE 2: f32 channel-major * mask       (output projection)
template <int MODE>
__global__ __launch_bounds__(128) void gemm512_kernel(
    const _Float16* __restrict__ act, const _Float16* __restrict__ wgt,
    const float* __restrict__ bias, void* __restrict__ outv,
    const float* __restrict__ maskr, float oscale) {
    __shared__ _Float16 lA[2][64][72]; // [buf][o rows][k=64], padded rows
    __shared__ _Float16 lB[2][64][72]; // [buf][p rows][k=64]
    const int b = blockIdx.z, oT = blockIdx.y * 64, pT = blockIdx.x * 64;
    const int t = threadIdx.x, lane = t & 31, wave = t >> 5;
    const int nl = lane & 15;
    const size_t actBase = ((size_t)b * 1024 + pT) * 512;

    auto stage = [&](int buf, int k0) {
        int row = t >> 1, cs = (t & 1) * 32; // each thread: 64B of A + 64B of B
        const _Float16* ga = wgt + (size_t)(oT + row) * 512 + k0 + cs;
        const _Float16* gb = act + actBase + (size_t)row * 512 + k0 + cs;
        _Float16* da = &lA[buf][row][cs];
        _Float16* db = &lB[buf][row][cs];
#pragma unroll
        for (int i = 0; i < 4; ++i) {
            async_cp16(da + i * 8, ga + i * 8);
            async_cp16(db + i * 8, gb + i * 8);
        }
    };

    v8f acc[4] = {};
    int buf = 0;
    stage(0, 0);
    async_wait0();
    __syncthreads();
    for (int k0 = 0; k0 < 512; k0 += 64) {
        if (k0 + 64 < 512) stage(buf ^ 1, k0 + 64); // prefetch next K-slab (async)
#pragma unroll
        for (int half = 0; half < 2; ++half) {
            const int kk = half * 32;
            const int ar = wave * 16 + nl, kc = (lane >> 4) * 8;
            v8h alo = *(const v8h*)&lA[buf][ar][kk + kc];
            v8h ahi = *(const v8h*)&lA[buf][ar][kk + kc + 16];
            v16h a = cat16(alo, ahi);
            const int kb = (lane >> 4) * 8;
#pragma unroll
            for (int j = 0; j < 4; ++j) {
                int br = j * 16 + nl;
                v8h blo = *(const v8h*)&lB[buf][br][kk + kb];
                v8h bhi = *(const v8h*)&lB[buf][br][kk + kb + 16];
                acc[j] = wmma_f16(a, cat16(blo, bhi), acc[j]);
            }
        }
        async_wait0();
        __syncthreads();
        buf ^= 1;
    }

    const int hi8 = (lane & 16) ? 8 : 0;
#pragma unroll
    for (int j = 0; j < 4; ++j) {
#pragma unroll
        for (int r = 0; r < 8; ++r) {
            int o = oT + wave * 16 + r + hi8;
            int p = pT + j * 16 + nl;
            float val = (acc[j][r] + bias[o]) * oscale;
            if (MODE == 0)
                ((_Float16*)outv)[((size_t)b * 1024 + p) * 512 + o] = (_Float16)val;
            else if (MODE == 1)
                ((_Float16*)outv)[((size_t)b * 512 + o) * 1024 + p] = (_Float16)val;
            else
                ((float*)outv)[((size_t)b * 512 + o) * 1024 + p] = val * maskr[b * 1024 + p];
        }
    }
}

// ---- flash attention: per (b,h), 1024 q x 1024 k, hd=64; scale pre-folded into q ----
__global__ __launch_bounds__(128) void attn_kernel(
    const _Float16* __restrict__ q, const _Float16* __restrict__ k,
    const _Float16* __restrict__ vt, _Float16* __restrict__ ao) {
    __shared__ _Float16 lK[2][32][72];    // [buf][key][d]
    __shared__ _Float16 lV[2][64][40];    // [buf][d][key]
    __shared__ _Float16 lP[4][16][40];    // per-wave P tile [m][key]
    const int bh = blockIdx.y, b = bh >> 3, h = bh & 7;
    const int t = threadIdx.x, lane = t & 31, wave = t >> 5;
    const int q0 = blockIdx.x * 64 + wave * 16;
    const int hi8 = (lane & 16) ? 8 : 0, nl = lane & 15;

    auto stage = [&](int buf, int kbase) {
        {
            int row = t >> 2, cs = (t & 3) * 16;
            const _Float16* g = k + ((size_t)b * 1024 + kbase + row) * 512 + h * 64 + cs;
            _Float16* d = &lK[buf][row][cs];
            async_cp16(d, g);
            async_cp16(d + 8, g + 8);
        }
        {
            int row = t >> 1, cs = (t & 1) * 16;
            const _Float16* g = vt + ((size_t)b * 512 + h * 64 + row) * 1024 + kbase + cs;
            _Float16* d = &lV[buf][row][cs];
            async_cp16(d, g);
            async_cp16(d + 8, g + 8);
        }
    };

    // Q fragments in A-matrix layout, register-resident for the whole kernel
    v16h qf[2];
    {
        int sr = q0 + nl;
        const _Float16* qp = q + ((size_t)b * 1024 + sr) * 512 + h * 64;
#pragma unroll
        for (int f = 0; f < 2; ++f) {
            int kc = f * 32 + (lane >> 4) * 8;
            v8h lo = *(const v8h*)(qp + kc);
            v8h hi = *(const v8h*)(qp + kc + 16);
            qf[f] = cat16(lo, hi);
        }
    }

    float mx[8], l[8];
#pragma unroll
    for (int r = 0; r < 8; ++r) { mx[r] = -INFINITY; l[r] = 0.0f; }
    v8f oacc[4] = {};

    stage(0, 0);
    async_wait0();
    __syncthreads();
    for (int kt = 0; kt < 32; ++kt) {
        const int buf = kt & 1;
        if (kt < 31) stage(buf ^ 1, (kt + 1) * 32); // async prefetch next tile

        // S = Q * K^T (16 q-rows x 32 keys), scale already in Q
        v8f s[2];
#pragma unroll
        for (int n = 0; n < 2; ++n) {
            int kr = n * 16 + nl, kc = (lane >> 4) * 8;
            v8h b0 = *(const v8h*)&lK[buf][kr][kc];
            v8h b1 = *(const v8h*)&lK[buf][kr][kc + 16];
            v8h c0 = *(const v8h*)&lK[buf][kr][32 + kc];
            v8h c1 = *(const v8h*)&lK[buf][kr][32 + kc + 16];
            v8f z = {};
            z = wmma_f16(qf[0], cat16(b0, b1), z);
            z = wmma_f16(qf[1], cat16(c0, c1), z);
            s[n] = z;
        }

        // online softmax: only the row-max needs a cross-lane reduction per tile;
        // the row-sum is kept as a per-lane partial and reduced once at the end.
        float corr[8];
#pragma unroll
        for (int r = 0; r < 8; ++r) {
            float t0 = s[0][r];
            float t1 = s[1][r];
            float mt = redmax16(fmaxf(t0, t1));
            float mnew = fmaxf(mx[r], mt);
            float c = __expf(mx[r] - mnew);
            float p0 = __expf(t0 - mnew);
            float p1 = __expf(t1 - mnew);
            l[r] = l[r] * c + (p0 + p1); // per-lane partial
            mx[r] = mnew;
            corr[r] = c;
            lP[wave][r + hi8][nl]      = (_Float16)p0;
            lP[wave][r + hi8][nl + 16] = (_Float16)p1;
        }
#pragma unroll
        for (int j = 0; j < 4; ++j)
#pragma unroll
            for (int r = 0; r < 8; ++r) oacc[j][r] *= corr[r];

        // O += P * V
        v16h pf;
        {
            int kc = (lane >> 4) * 8;
            v8h lo = *(const v8h*)&lP[wave][nl][kc];
            v8h hi = *(const v8h*)&lP[wave][nl][kc + 16];
            pf = cat16(lo, hi);
        }
#pragma unroll
        for (int j = 0; j < 4; ++j) {
            int dr = j * 16 + nl, kc = (lane >> 4) * 8;
            v8h b0 = *(const v8h*)&lV[buf][dr][kc];
            v8h b1 = *(const v8h*)&lV[buf][dr][kc + 16];
            oacc[j] = wmma_f16(pf, cat16(b0, b1), oacc[j]);
        }

        async_wait0();   // next tile resident before anyone proceeds
        __syncthreads(); // single barrier per iteration (double-buffered)
    }

    // final lane-reduction of the row sums, normalize, store f16 [b][s][512]
#pragma unroll
    for (int r = 0; r < 8; ++r) l[r] = redsum16(l[r]);
#pragma unroll
    for (int j = 0; j < 4; ++j) {
#pragma unroll
        for (int r = 0; r < 8; ++r) {
            int s_ = q0 + r + hi8;
            float inv = 1.0f / l[r];
            ao[((size_t)b * 1024 + s_) * 512 + h * 64 + j * 16 + nl] =
                (_Float16)(oacc[j][r] * inv);
        }
    }
}

// ---- bilinear 2x upsample (align_corners=False) [b][c][32][32] -> [b][c][64][64] ----
__global__ void upsample2_kernel(const float* __restrict__ proj, float* __restrict__ out) {
    int idx = blockIdx.x * blockDim.x + threadIdx.x;
    if (idx >= 8 * 512 * 64 * 64) return;
    int j = idx & 63, i = (idx >> 6) & 63, c = (idx >> 12) & 511, b = idx >> 21;
    const float* P = proj + ((size_t)b * 512 + c) * 1024;
    int ry = i >> 1, rx = j >> 1;
    int y0, y1, x0, x1;
    float wy0, wy1, wx0, wx1;
    if (i & 1) { y0 = ry; y1 = ry + 1; if (y1 > 31) y1 = 31; wy0 = 0.75f; wy1 = 0.25f; }
    else       { y0 = ry - 1; if (y0 < 0) y0 = 0; y1 = ry;   wy0 = 0.25f; wy1 = 0.75f; }
    if (j & 1) { x0 = rx; x1 = rx + 1; if (x1 > 31) x1 = 31; wx0 = 0.75f; wx1 = 0.25f; }
    else       { x0 = rx - 1; if (x0 < 0) x0 = 0; x1 = rx;   wx0 = 0.25f; wx1 = 0.75f; }
    float v = wy0 * (wx0 * P[y0 * 32 + x0] + wx1 * P[y0 * 32 + x1]) +
              wy1 * (wx0 * P[y1 * 32 + x0] + wx1 * P[y1 * 32 + x1]);
    out[idx] = v;
}

extern "C" void kernel_launch(void* const* d_in, const int* in_sizes, int n_in,
                              void* d_out, int out_size, void* d_ws, size_t ws_size,
                              hipStream_t stream) {
    const float* x  = (const float*)d_in[0];
    const float* z  = (const float*)d_in[1];
    const float* m  = (const float*)d_in[2];
    const float* wq = (const float*)d_in[3];
    const float* bq = (const float*)d_in[4];
    const float* wk = (const float*)d_in[5];
    const float* bk = (const float*)d_in[6];
    const float* wv = (const float*)d_in[7];
    const float* bv = (const float*)d_in[8];
    const float* wo = (const float*)d_in[9];
    const float* bo = (const float*)d_in[10];

    char* ws = (char*)d_ws;
    _Float16* xr   = (_Float16*)(ws + OFF_XR);
    _Float16* zr   = (_Float16*)(ws + OFF_ZR);
    _Float16* qh   = (_Float16*)(ws + OFF_Q);
    _Float16* kh   = (_Float16*)(ws + OFF_K);
    _Float16* vth  = (_Float16*)(ws + OFF_VT);
    _Float16* aoh  = (_Float16*)(ws + OFF_AO);
    float*    proj = (float*)(ws + OFF_PROJ);
    float*    mr   = (float*)(ws + OFF_MR);
    _Float16* w4   = (_Float16*)(ws + OFF_W);
    _Float16* wqh = w4;
    _Float16* wkh = w4 + 262144;
    _Float16* wvh = w4 + 2 * 262144;
    _Float16* woh = w4 + 3 * 262144;

    pool_f16_kernel<<<16384, 256, 0, stream>>>(x, xr);
    pool_f16_kernel<<<16384, 256, 0, stream>>>(z, zr);
    pool_mask_kernel<<<32, 256, 0, stream>>>(m, mr);
    cvt_w_kernel<<<4096, 256, 0, stream>>>(wq, wk, wv, wo, w4);

    dim3 ggrid(16, 8, 8); // p-tiles, o-tiles, batch
    gemm512_kernel<0><<<ggrid, 128, 0, stream>>>(xr, wqh, bq, qh, nullptr, SCALE);
    gemm512_kernel<0><<<ggrid, 128, 0, stream>>>(zr, wkh, bk, kh, nullptr, 1.0f);
    gemm512_kernel<1><<<ggrid, 128, 0, stream>>>(zr, wvh, bv, vth, nullptr, 1.0f);

    attn_kernel<<<dim3(16, 64), 128, 0, stream>>>(qh, kh, vth, aoh);

    gemm512_kernel<2><<<ggrid, 128, 0, stream>>>(aoh, woh, bo, proj, mr, 1.0f);
    upsample2_kernel<<<65536, 256, 0, stream>>>(proj, (float*)d_out);
}
